// LLamaAttentionBlock_7834020348284
// MI455X (gfx1250) — matrix operand order repaired and checked by reading, not compiled, not run on hardware
//
#include <hip/hip_runtime.h>
#include <hip/hip_bf16.h>

typedef __attribute__((ext_vector_type(16))) _Float16 v16h;
typedef __attribute__((ext_vector_type(8)))  _Float16 v8h;
typedef __attribute__((ext_vector_type(8)))  float    v8f;

#define Bn     8
#define Hn     32
#define QL     128
#define Dd     128
#define CACHE  1920
#define EMB    2048
#define CK     64      // keys per LDS chunk
#define KP     136     // K LDS pitch (f16), padded vs 128 to spread banks
#define VP     72      // V^T LDS pitch (f16), padded vs 64
#define PP     72      // P LDS pitch (f16), padded vs 64

__global__ __launch_bounds__(256, 1)
void fa_kernel(const float* __restrict__ Qg, const float* __restrict__ Kg,
               const float* __restrict__ Vg, const float* __restrict__ Kc,
               const float* __restrict__ Vc, const float* __restrict__ w1,
               const float* __restrict__ w2, const float* __restrict__ w3,
               float* __restrict__ Og)
{
    __shared__ __align__(16) _Float16 sK[CK * KP];        // [key][dim]  (* w2)
    __shared__ __align__(16) _Float16 sV[Dd * VP];        // [dim][key]  (transposed)
    __shared__ __align__(16) _Float16 sP[8 * 16 * PP];    // per-wave 16x64 P tile

    const int bh   = blockIdx.x;          // (b*H + h), 0..255
    const int tid  = threadIdx.x;
    const int lane = tid & 31;
    const int wv   = tid >> 5;            // wave 0..7 -> query rows wv*16..wv*16+15
    const int col  = lane & 15;
    const int hv   = lane >> 4;           // lane half
    const float scl = 0.08838834764831845f;   // 1/sqrt(128)

    // ---- Q A-fragments: 16x32 per kk, folded scale w1*C, contiguous k = 16h + s ----
    const float* qrow = Qg + ((size_t)bh * QL + (wv * 16 + col)) * Dd;
    v16h qa[4];
#pragma unroll
    for (int kk = 0; kk < 4; ++kk) {
        const int base = kk * 32 + hv * 16;
#pragma unroll
        for (int j = 0; j < 8; ++j) {
            qa[kk][j]     = (_Float16)(qrow[base + j]      * w1[base + j]      * scl);
            qa[kk][8 + j] = (_Float16)(qrow[base + 8 + j] * w1[base + 8 + j] * scl);
        }
    }

    float mrow[8], lrow[8];
    v8f   acco[8];
#pragma unroll
    for (int r = 0; r < 8; ++r) { mrow[r] = -__builtin_inff(); lrow[r] = 0.f; }
#pragma unroll
    for (int ot = 0; ot < 8; ++ot) acco[ot] = (v8f){};

    const int qbase = wv * 16 + hv * 8;   // first query row tracked by this lane

    for (int kb = 0; kb < EMB; kb += CK) {
        __syncthreads();                  // previous chunk fully consumed
        // ---- cooperative stage: K*w2 row-major, V transposed, fp32->f16 ----
#pragma unroll
        for (int s = 0; s < 8; ++s) {
            const int linear = tid + 256 * s;     // float4 tiles of the 64x128 chunk
            const int row = linear >> 5;          // key within chunk
            const int d4  = (linear & 31) * 4;    // dim start
            const int kg  = kb + row;
            const float *ksrc, *vsrc;
            if (kg < CACHE) {
                const size_t off = ((size_t)bh * CACHE + kg) * Dd + d4;
                ksrc = Kc + off; vsrc = Vc + off;
            } else {
                const size_t off = ((size_t)bh * QL + (kg - CACHE)) * Dd + d4;
                ksrc = Kg + off; vsrc = Vg + off;
            }
            __builtin_prefetch(ksrc + CK * Dd, 0, 1);   // next chunk, speculative
            const float4 kq = *(const float4*)ksrc;
            const float4 vq = *(const float4*)vsrc;
            _Float16* kd = &sK[row * KP + d4];
            kd[0] = (_Float16)(kq.x * w2[d4 + 0]);
            kd[1] = (_Float16)(kq.y * w2[d4 + 1]);
            kd[2] = (_Float16)(kq.z * w2[d4 + 2]);
            kd[3] = (_Float16)(kq.w * w2[d4 + 3]);
            sV[(d4 + 0) * VP + row] = (_Float16)vq.x;
            sV[(d4 + 1) * VP + row] = (_Float16)vq.y;
            sV[(d4 + 2) * VP + row] = (_Float16)vq.z;
            sV[(d4 + 3) * VP + row] = (_Float16)vq.w;
        }
        __syncthreads();

        // ---- S = Qhat @ Khat^T : 4 key-tiles x 4 K-steps = 16 WMMAs ----
        v8f sacc[4];
#pragma unroll
        for (int nt = 0; nt < 4; ++nt) {
            v8f c = (v8f){};
            const _Float16* kb16 = &sK[(nt * 16 + col) * KP];   // B col = key
#pragma unroll
            for (int kk = 0; kk < 4; ++kk) {
                const v16h bf = *(const v16h*)(kb16 + kk * 32 + hv * 16);
                c = __builtin_amdgcn_wmma_f32_16x16x32_f16(false, qa[kk], false, bf,
                                                           (short)0, c, false, false);
            }
            sacc[nt] = c;
        }

        // ---- w3 scale, causal mask, chunk row-max ----
        float cm[8];
#pragma unroll
        for (int r = 0; r < 8; ++r) cm[r] = -__builtin_inff();
#pragma unroll
        for (int nt = 0; nt < 4; ++nt) {
            const int   kidx = kb + nt * 16 + col;
            const float w3v  = w3[kidx];
            const int   rel  = kidx - CACHE;
#pragma unroll
            for (int r = 0; r < 8; ++r) {
                float sx = sacc[nt][r] * w3v;
                sx = (rel <= qbase + r) ? sx : -__builtin_inff();
                sacc[nt][r] = sx;
                cm[r] = fmaxf(cm[r], sx);
            }
        }
#pragma unroll
        for (int msk = 1; msk <= 8; msk <<= 1)
#pragma unroll
            for (int r = 0; r < 8; ++r)
                cm[r] = fmaxf(cm[r], __shfl_xor(cm[r], msk, 32));

        // ---- online softmax rescale ----
        float alpha[8];
#pragma unroll
        for (int r = 0; r < 8; ++r) {
            const float mn = fmaxf(mrow[r], cm[r]);
            alpha[r] = __expf(mrow[r] - mn);
            mrow[r]  = mn;
            lrow[r] *= alpha[r];
        }
#pragma unroll
        for (int ot = 0; ot < 8; ++ot)
#pragma unroll
            for (int r = 0; r < 8; ++r)
                acco[ot][r] *= alpha[r];

        // ---- P = exp(S - m): accumulate l, stage to LDS (C-layout -> row major)
        float psum[8];
#pragma unroll
        for (int r = 0; r < 8; ++r) psum[r] = 0.f;
        _Float16* prow = &sP[(wv * 16 + hv * 8) * PP + col];
#pragma unroll
        for (int nt = 0; nt < 4; ++nt)
#pragma unroll
            for (int r = 0; r < 8; ++r) {
                const float p = __expf(sacc[nt][r] - mrow[r]);
                psum[r] += p;
                prow[r * PP + nt * 16] = (_Float16)p;
            }
#pragma unroll
        for (int msk = 1; msk <= 8; msk <<= 1)
#pragma unroll
            for (int r = 0; r < 8; ++r)
                psum[r] += __shfl_xor(psum[r], msk, 32);
#pragma unroll
        for (int r = 0; r < 8; ++r) lrow[r] += psum[r];

        __syncthreads();                 // P tiles visible for A-layout reload

        // ---- O += P @ V : 2 A-frags x 8 dim-tiles = 16 WMMAs ----
        const _Float16* pb = &sP[(wv * 16 + col) * PP];   // A row = query col
#pragma unroll
        for (int k2 = 0; k2 < 2; ++k2) {
            v16h pa;
            const v8h lo = *(const v8h*)(pb + k2 * 32 + hv * 16);
            const v8h hi = *(const v8h*)(pb + k2 * 32 + hv * 16 + 8);
#pragma unroll
            for (int j = 0; j < 8; ++j) { pa[j] = lo[j]; pa[8 + j] = hi[j]; }
#pragma unroll
            for (int ot = 0; ot < 8; ++ot) {
                const v16h vb = *(const v16h*)(&sV[(ot * 16 + col) * VP + k2 * 32 + hv * 16]);
                acco[ot] = __builtin_amdgcn_wmma_f32_16x16x32_f16(false, pa, false, vb,
                                                                  (short)0, acco[ot],
                                                                  false, false);
            }
        }
    }

    // ---- normalize and store ----
#pragma unroll
    for (int r = 0; r < 8; ++r) {
        const float inv = 1.0f / lrow[r];
        float* op = Og + ((size_t)bh * QL + qbase + r) * Dd + col;
#pragma unroll
        for (int ot = 0; ot < 8; ++ot)
            op[ot * 16] = acco[ot][r] * inv;
    }
}

extern "C" void kernel_launch(void* const* d_in, const int* in_sizes, int n_in,
                              void* d_out, int out_size, void* d_ws, size_t ws_size,
                              hipStream_t stream) {
    const float* Q  = (const float*)d_in[0];
    const float* K  = (const float*)d_in[1];
    const float* V  = (const float*)d_in[2];
    const float* Kc = (const float*)d_in[3];
    const float* Vc = (const float*)d_in[4];
    const float* w1 = (const float*)d_in[5];
    const float* w2 = (const float*)d_in[6];
    const float* w3 = (const float*)d_in[7];
    float* out = (float*)d_out;
    dim3 grid(Bn * Hn);   // one workgroup per (b,h)
    dim3 block(256);      // 8 wave32 waves, 16 query rows each
    hipLaunchKernelGGL(fa_kernel, grid, block, 0, stream,
                       Q, K, V, Kc, Vc, w1, w2, w3, out);
}
